// Mamba2_31851477467737
// MI455X (gfx1250) — matrix operand order repaired and checked by reading, not compiled, not run on hardware
//
#include <hip/hip_runtime.h>

// ---------------- problem constants ----------------
#define BATCH   16
#define LSEQ    1024
#define DM      768
#define D_INNER 1536
#define HEADDIM 64
#define NHEADS  24
#define DSTATE  64
#define CONV_DIM (D_INNER + 2*DSTATE)            // 1664
#define DIN      (2*D_INNER + 2*DSTATE + NHEADS) // 3224
#define ROWS    (BATCH*LSEQ)                     // 16384
#define IMGH    32
#define IMGW    32

typedef __bf16 bf16_t;
typedef __attribute__((ext_vector_type(16))) __bf16 v16bf;
typedef __attribute__((ext_vector_type(8)))  __bf16 v8bf;
typedef __attribute__((ext_vector_type(8)))  float  v8f;

union frag16 { v16bf v; v8bf h[2]; };

// ============================================================
// fp32 -> bf16 bulk conversion (n must be a multiple of 8; it is)
// ============================================================
__global__ __launch_bounds__(256)
void cvt_f32_bf16(const float* __restrict__ in, bf16_t* __restrict__ out, int n) {
  int i = (blockIdx.x * 256 + threadIdx.x) * 8;
  if (i + 8 > n) return;
  float4 a = *(const float4*)(in + i);
  float4 b = *(const float4*)(in + i + 4);
  v8bf r;
  r[0] = (bf16_t)a.x; r[1] = (bf16_t)a.y; r[2] = (bf16_t)a.z; r[3] = (bf16_t)a.w;
  r[4] = (bf16_t)b.x; r[5] = (bf16_t)b.y; r[6] = (bf16_t)b.z; r[7] = (bf16_t)b.w;
  *(v8bf*)(out + i) = r;
}

// ============================================================
// Tiled GEMM (compile-time shapes): C[M,N] = A[M,K] * B[K,N]
// A,B bf16, C fp32. Block 256 thr (8 waves). Tile 256(M) x 64(N), BK=32.
// Each wave owns 32(M) x 64(N): 8 WMMA accumulators; B fragments reused
// across both M sub-tiles (4 b128 A + 64 16-bit B ds_loads : 8 wmma per K-step per wave).
// Double-buffered LDS (40KB), one barrier per K-step, loads overlap WMMA.
// Requires M%256==0, K%32==0; N tail handled block-uniformly.
// ============================================================
#define BM 256
#define BN 64
#define BK 32

template<int M, int N, int K, int LDA, int LDB, int LDC>
__global__ __launch_bounds__(256)
void gemm_bf16wmma(const bf16_t* __restrict__ A, const bf16_t* __restrict__ B,
                   float* __restrict__ C) {
  __shared__ bf16_t As[2][BM][BK];   // 2 x 16 KB
  __shared__ bf16_t Bs[2][BK][BN];   // 2 x  4 KB

  const int tid  = threadIdx.x;
  const int n0   = blockIdx.x * BN;
  const int m0   = blockIdx.y * BM;
  const int wave = tid >> 5;
  const int lane = tid & 31;
  const int g    = lane >> 4;        // half-wave group
  const int ml   = lane & 15;
  const int wm   = wave * 32;        // wave M offset in tile
  const bool fullN = (n0 + BN <= N); // block-uniform (constant for N%64==0)

  // A staging: each thread stages one full 32-element row (4 x b128)
  const int ar = tid;                // row 0..255
  // B staging: 256 16B-chunks / 256 thr = 1 each
  const int br = tid >> 3;           // row 0..31
  const int bo = (tid & 7) * 8;      // col offset 0..56

  const bf16_t* Ap = A + (size_t)(m0 + ar) * LDA;
  const bf16_t* Bp = B + (size_t)br * LDB + n0 + bo;

  v8f acc[2][4] = {};
  v8bf a0, a1, a2, a3, breg;

  // ---- prologue: stage tile 0 ----
  a0 = *(const v8bf*)(Ap);
  a1 = *(const v8bf*)(Ap + 8);
  a2 = *(const v8bf*)(Ap + 16);
  a3 = *(const v8bf*)(Ap + 24);
  if (fullN) {
    breg = *(const v8bf*)(Bp);
  } else {
#pragma unroll
    for (int e = 0; e < 8; ++e) {
      int n = n0 + bo + e;
      breg[e] = (n < N) ? B[(size_t)br * LDB + n] : (bf16_t)0.0f;
    }
  }
  *(v8bf*)&As[0][ar][0]  = a0;
  *(v8bf*)&As[0][ar][8]  = a1;
  *(v8bf*)&As[0][ar][16] = a2;
  *(v8bf*)&As[0][ar][24] = a3;
  *(v8bf*)&Bs[0][br][bo] = breg;
  __syncthreads();

  constexpr int KT = K / BK;
  for (int kt = 0; kt < KT; ++kt) {
    const int cur = kt & 1;
    const bool more = (kt + 1 < KT);

    // ---- issue global loads for next tile (in flight during WMMA) ----
    if (more) {
      const int k1 = (kt + 1) * BK;
      a0 = *(const v8bf*)(Ap + k1);
      a1 = *(const v8bf*)(Ap + k1 + 8);
      a2 = *(const v8bf*)(Ap + k1 + 16);
      a3 = *(const v8bf*)(Ap + k1 + 24);
      if (fullN) {
        breg = *(const v8bf*)(Bp + (size_t)(kt + 1) * BK * LDB);
      } else {
#pragma unroll
        for (int e = 0; e < 8; ++e) {
          int n = n0 + bo + e;
          breg[e] = (n < N) ? B[(size_t)(k1 + br) * LDB + n] : (bf16_t)0.0f;
        }
      }
    }

    // ---- compute: load 4 B fragments once, reuse over 2 M sub-tiles ----
    frag16 bfr[4];
#pragma unroll
    for (int j = 0; j < 4; ++j) {
#pragma unroll
      for (int s = 0; s < 16; ++s)
        bfr[j].v[s] = Bs[cur][(s < 8 ? g*8 + s : 16 + g*8 + (s - 8))][j*16 + ml];
    }
#pragma unroll
    for (int i = 0; i < 2; ++i) {
      frag16 ua;
      const bf16_t* Arow = &As[cur][wm + i*16 + ml][0];
      ua.h[0] = *(const v8bf*)(Arow + g*8);
      ua.h[1] = *(const v8bf*)(Arow + 16 + g*8);
#pragma unroll
      for (int j = 0; j < 4; ++j) {
        acc[i][j] = __builtin_amdgcn_wmma_f32_16x16x32_bf16(
            false, ua.v, false, bfr[j].v, (short)0, acc[i][j], false, false);
      }
    }

    // ---- stage next tile into the other buffer ----
    if (more) {
      const int nb = cur ^ 1;
      *(v8bf*)&As[nb][ar][0]  = a0;
      *(v8bf*)&As[nb][ar][8]  = a1;
      *(v8bf*)&As[nb][ar][16] = a2;
      *(v8bf*)&As[nb][ar][24] = a3;
      *(v8bf*)&Bs[nb][br][bo] = breg;
    }
    __syncthreads();
  }

  // ---- store C (fp32); LDC is a compile-time constant ----
#pragma unroll
  for (int i = 0; i < 2; ++i)
#pragma unroll
    for (int j = 0; j < 4; ++j)
#pragma unroll
      for (int r = 0; r < 8; ++r) {
        int mm = m0 + wm + i*16 + r + g*8;
        int nn = n0 + j*16 + ml;
        if (fullN || nn < N) C[(size_t)mm * LDC + nn] = acc[i][j][r];
      }
}

// ============================================================
// dA[row,h] = softplus(proj[row, D_INNER+CONV_DIM+h] + dt_bias[h]) * exp(A_log[h])
// ============================================================
__global__ __launch_bounds__(256)
void da_kernel(const float* __restrict__ proj, const float* __restrict__ dt_bias,
               const float* __restrict__ A_log, float* __restrict__ dA) {
  int idx = blockIdx.x * 256 + threadIdx.x;
  if (idx >= ROWS * NHEADS) return;
  int h = idx % NHEADS;
  int row = idx / NHEADS;
  float x = proj[(size_t)row * DIN + D_INNER + CONV_DIM + h] + dt_bias[h];
  float sp = (x > 20.0f) ? x : log1pf(__expf(x));
  dA[idx] = sp * __expf(A_log[h]);
}

// ============================================================
// Depthwise 3x3 conv (SAME, zero pad) + bias + SiLU on the xBC slice of
// proj; split into x (fp32), Vs = x*dA (bf16), BmT (bf16, [b][d][s]!),
// Cm (bf16, [b][s][d]).  One block per (b, l); threads loop channels.
// ============================================================
__global__ __launch_bounds__(256)
void conv_silu_split(const float* __restrict__ proj, const float* __restrict__ kw,
                     const float* __restrict__ kb, const float* __restrict__ dA,
                     float* __restrict__ xs, bf16_t* __restrict__ Vs,
                     bf16_t* __restrict__ BmT, bf16_t* __restrict__ Cmb) {
  const int bl = blockIdx.x;
  const int b = bl / LSEQ;
  const int l = bl % LSEQ;
  const int py = l >> 5;
  const int px = l & 31;
  const size_t row = (size_t)bl;

  for (int c = threadIdx.x; c < CONV_DIM; c += 256) {
    float s = kb[c];
#pragma unroll
    for (int ky = 0; ky < 3; ++ky) {
#pragma unroll
      for (int kx = 0; kx < 3; ++kx) {
        int yy = py + ky - 1, xx = px + kx - 1;
        if (yy >= 0 && yy < IMGH && xx >= 0 && xx < IMGW) {
          size_t r2 = (size_t)b * LSEQ + yy * IMGW + xx;
          s += proj[r2 * DIN + D_INNER + c] * kw[(ky*3 + kx) * CONV_DIM + c];
        }
      }
    }
    float v = s / (1.0f + __expf(-s));   // SiLU
    if (c < D_INNER) {
      xs[row * D_INNER + c] = v;
      int h = c >> 6;
      float da = dA[row * NHEADS + h];
      Vs[row * D_INNER + c] = (bf16_t)(v * da);
    } else if (c < D_INNER + DSTATE) {
      int d = c - D_INNER;
      BmT[((size_t)b * DSTATE + d) * LSEQ + l] = (bf16_t)v;   // transposed
    } else {
      Cmb[row * DSTATE + (c - D_INNER - DSTATE)] = (bf16_t)v;
    }
  }
}

// ============================================================
// KV[b,h,d,v] = sum_s BmT[b,d,s] * Vs[b,s,h,v]   (64x64 per (b,h))
// One block (4 waves) per (b,h). Wave w owns d rows [16w,16w+16).
// ============================================================
__global__ __launch_bounds__(128)
void kv_kernel(const bf16_t* __restrict__ BmT, const bf16_t* __restrict__ Vs,
               bf16_t* __restrict__ KV) {
  const int bh = blockIdx.x;
  const int b  = bh / NHEADS;
  const int h  = bh % NHEADS;
  const int wave = threadIdx.x >> 5;
  const int lane = threadIdx.x & 31;
  const int g  = lane >> 4;
  const int ml = lane & 15;
  const int d0 = wave * 16;

  v8f acc[4] = {};

  for (int s0 = 0; s0 < LSEQ; s0 += 32) {
    frag16 ua;  // A[m][K] = BmT[b][d0+m][s0+K] : contiguous chunks
    const bf16_t* arow = BmT + ((size_t)b * DSTATE + d0 + ml) * LSEQ + s0;
    ua.h[0] = *(const v8bf*)(arow + g*8);
    ua.h[1] = *(const v8bf*)(arow + 16 + g*8);

    const bf16_t* vrow = Vs + ((size_t)b * LSEQ + s0) * D_INNER + h * HEADDIM + ml;
#pragma unroll
    for (int j = 0; j < 4; ++j) {
      frag16 ub;   // lane holds column v = 16*j + lane%16, elements = k (table order)
#pragma unroll
      for (int s = 0; s < 16; ++s)
        ub.v[s] = vrow[(size_t)(s < 8 ? g*8 + s : 16 + g*8 + (s - 8)) * D_INNER + 16*j];
      acc[j] = __builtin_amdgcn_wmma_f32_16x16x32_bf16(
          false, ua.v, false, ub.v, (short)0, acc[j], false, false);
    }
  }

#pragma unroll
  for (int j = 0; j < 4; ++j)
#pragma unroll
    for (int r = 0; r < 8; ++r) {
      int d = d0 + r + g*8;
      int v = 16*j + ml;
      KV[((size_t)bh * DSTATE + d) * HEADDIM + v] = (bf16_t)acc[j][r];
    }
}

// ============================================================
// y[b,s,h,:] = Cm[b,s,:] @ KV[b,h,:,:] + x*D_skip  -> yw fp32 [ROWS, D_INNER]
// One block (8 waves) per (b,h); each wave loops over s-tiles of 16.
// ============================================================
__global__ __launch_bounds__(256)
void y_kernel(const bf16_t* __restrict__ Cmb, const bf16_t* __restrict__ KV,
              const float* __restrict__ xs, const float* __restrict__ Dskip,
              float* __restrict__ yw) {
  const int bh = blockIdx.x;
  const int b  = bh / NHEADS;
  const int h  = bh % NHEADS;
  const int wave = threadIdx.x >> 5;
  const int lane = threadIdx.x & 31;
  const int g  = lane >> 4;
  const int ml = lane & 15;
  const float dsk = Dskip[h];

  for (int t = wave; t < LSEQ/16; t += 8) {
    const int s0 = t * 16;
    v8f acc[4] = {};
#pragma unroll
    for (int k0 = 0; k0 < DSTATE; k0 += 32) {
      frag16 ua;  // A[m][K] = Cm[s0+m][k0+K] : contiguous chunks
      const bf16_t* crow = Cmb + ((size_t)b * LSEQ + s0 + ml) * DSTATE + k0;
      ua.h[0] = *(const v8bf*)(crow + g*8);
      ua.h[1] = *(const v8bf*)(crow + 16 + g*8);
#pragma unroll
      for (int j = 0; j < 4; ++j) {
        frag16 ub;  // lane holds column 16*j + lane%16 of KV, elements = k (table order)
        const bf16_t* krow = KV + ((size_t)bh * DSTATE + k0) * HEADDIM + 16*j + ml;
#pragma unroll
        for (int s = 0; s < 16; ++s)
          ub.v[s] = krow[(s < 8 ? g*8 + s : 16 + g*8 + (s - 8)) * HEADDIM];
        acc[j] = __builtin_amdgcn_wmma_f32_16x16x32_bf16(
            false, ua.v, false, ub.v, (short)0, acc[j], false, false);
      }
    }
#pragma unroll
    for (int j = 0; j < 4; ++j)
#pragma unroll
      for (int r = 0; r < 8; ++r) {
        int s = s0 + r + g*8;
        int col = h * HEADDIM + 16*j + ml;
        size_t row = (size_t)b * LSEQ + s;
        float xv = xs[row * D_INNER + col];
        yw[row * D_INNER + col] = acc[j][r] + xv * dsk;
      }
  }
}

// ============================================================
// LayerNorm over D_INNER + gate by z (= proj[...,:D_INNER]) -> yg (bf16)
// One block of 256 per row.
// ============================================================
__global__ __launch_bounds__(256)
void ln_gate(const float* __restrict__ yw, const float* __restrict__ proj,
             const float* __restrict__ scale, const float* __restrict__ bias,
             bf16_t* __restrict__ yg) {
  __shared__ float s1[256], s2[256];
  const size_t row = blockIdx.x;
  const int tid = threadIdx.x;

  float s = 0.f, sq = 0.f;
  for (int c = tid; c < D_INNER; c += 256) {
    float v = yw[row * D_INNER + c];
    s += v; sq += v * v;
  }
  s1[tid] = s; s2[tid] = sq;
  __syncthreads();
  for (int off = 128; off > 0; off >>= 1) {
    if (tid < off) { s1[tid] += s1[tid + off]; s2[tid] += s2[tid + off]; }
    __syncthreads();
  }
  const float mu = s1[0] * (1.0f / D_INNER);
  const float var = s2[0] * (1.0f / D_INNER) - mu * mu;
  const float rs = rsqrtf(var + 1e-6f);

  for (int c = tid; c < D_INNER; c += 256) {
    float v = (yw[row * D_INNER + c] - mu) * rs * scale[c] + bias[c];
    yg[row * D_INNER + c] = (bf16_t)(v * proj[row * DIN + c]);  // gate by z
  }
}

// ============================================================
// Host-side launcher
// ============================================================
extern "C" void kernel_launch(void* const* d_in, const int* in_sizes, int n_in,
                              void* d_out, int out_size, void* d_ws, size_t ws_size,
                              hipStream_t stream) {
  const float* u        = (const float*)d_in[0];
  const float* w_in     = (const float*)d_in[1];
  const float* convk    = (const float*)d_in[2];
  const float* convb    = (const float*)d_in[3];
  const float* dt_bias  = (const float*)d_in[4];
  const float* A_log    = (const float*)d_in[5];
  const float* D_skip   = (const float*)d_in[6];
  const float* ln_scale = (const float*)d_in[7];
  const float* ln_bias  = (const float*)d_in[8];
  const float* w_out    = (const float*)d_in[9];
  float* out = (float*)d_out;

  char* ws = (char*)d_ws;
  size_t off = 0;
  auto take = [&](size_t bytes) { char* p = ws + off; off += (bytes + 255) & ~(size_t)255; return p; };

  float*  proj  = (float*)take((size_t)ROWS * DIN * 4);
  float*  dA    = (float*)take((size_t)ROWS * NHEADS * 4);
  float*  xs    = (float*)take((size_t)ROWS * D_INNER * 4);
  bf16_t* Vs    = (bf16_t*)take((size_t)ROWS * D_INNER * 2);
  bf16_t* BmT   = (bf16_t*)take((size_t)BATCH * DSTATE * LSEQ * 2);
  bf16_t* Cmb   = (bf16_t*)take((size_t)ROWS * DSTATE * 2);
  bf16_t* KV    = (bf16_t*)take((size_t)BATCH * NHEADS * DSTATE * HEADDIM * 2);
  float*  yw    = (float*)take((size_t)ROWS * D_INNER * 4);
  bf16_t* yg    = (bf16_t*)take((size_t)ROWS * D_INNER * 2);
  bf16_t* u_bf  = (bf16_t*)take((size_t)ROWS * DM * 2);
  bf16_t* wi_bf = (bf16_t*)take((size_t)DM * DIN * 2);
  bf16_t* wo_bf = (bf16_t*)take((size_t)D_INNER * DM * 2);

  // 0) bf16 copies of GEMM operands
  cvt_f32_bf16<<<(ROWS*DM/8 + 255)/256, 256, 0, stream>>>(u, u_bf, ROWS*DM);
  cvt_f32_bf16<<<(DM*DIN/8 + 255)/256, 256, 0, stream>>>(w_in, wi_bf, DM*DIN);
  cvt_f32_bf16<<<(D_INNER*DM/8 + 255)/256, 256, 0, stream>>>(w_out, wo_bf, D_INNER*DM);

  // 1) proj = u @ w_in   [16384 x 3224]
  {
    dim3 grid((DIN + BN - 1) / BN, ROWS / BM);
    gemm_bf16wmma<ROWS, DIN, DM, DM, DIN, DIN>
        <<<grid, 256, 0, stream>>>(u_bf, wi_bf, proj);
  }
  // 2) dA
  da_kernel<<<(ROWS * NHEADS + 255) / 256, 256, 0, stream>>>(proj, dt_bias, A_log, dA);
  // 3) conv + SiLU + split
  conv_silu_split<<<ROWS, 256, 0, stream>>>(proj, convk, convb, dA, xs, Vs, BmT, Cmb);
  // 4) KV
  kv_kernel<<<BATCH * NHEADS, 128, 0, stream>>>(BmT, Vs, KV);
  // 5) y = Cm @ KV + x*D_skip
  y_kernel<<<BATCH * NHEADS, 256, 0, stream>>>(Cmb, KV, xs, D_skip, yw);
  // 6) LayerNorm + gate -> bf16
  ln_gate<<<ROWS, 256, 0, stream>>>(yw, proj, ln_scale, ln_bias, yg);
  // 7) out = yg @ w_out  [16384 x 768]
  {
    dim3 grid((DM + BN - 1) / BN, ROWS / BM);
    gemm_bf16wmma<ROWS, DM, D_INNER, D_INNER, DM, DM>
        <<<grid, 256, 0, stream>>>(yg, wo_bf, out);
  }
}
